// DBSSBlock_17325898072434
// MI455X (gfx1250) — matrix operand order repaired and checked by reading, not compiled, not run on hardware
//
#include <hip/hip_runtime.h>
#include <hip/hip_bf16.h>

// ---------------------------------------------------------------------------
// Types for CDNA5 WMMA
// ---------------------------------------------------------------------------
typedef __attribute__((ext_vector_type(16))) __bf16 v16bf;
typedef __attribute__((ext_vector_type(8)))  float  v8f;
typedef __attribute__((ext_vector_type(4)))  unsigned int uint4v;
typedef __attribute__((ext_vector_type(8)))  unsigned int uint8v;

union WmmaOp { v16bf v; unsigned u[8]; };

__device__ __forceinline__ unsigned bf16_bits(float x) {
    unsigned u = __float_as_uint(x);
    return (u + 0x7FFFu + ((u >> 16) & 1u)) >> 16;   // RNE, NaN-sloppy (fine for NN)
}

// snake index helpers (H = W = 64).  pos_r is an involution.
__device__ __forceinline__ int pos_r(int f) {
    int r = f >> 6, c = f & 63;
    return (r << 6) | ((r & 1) ? 63 - c : c);
}
__device__ __forceinline__ int idx_c_map(int p) {        // snake pos -> flat
    int i = p >> 6, j0 = p & 63;
    int j = (i & 1) ? 63 - j0 : j0;
    return (j << 6) | i;
}
__device__ __forceinline__ int pos_c(int f) {            // flat -> snake pos
    int r = f >> 6, c = f & 63;
    return (c << 6) | ((c & 1) ? 63 - r : r);
}

// ---------------------------------------------------------------------------
// Pack fp32 weights -> bf16 pairs (one dword = two consecutive Cin values)
// ---------------------------------------------------------------------------
__global__ void pack_bf16_kernel(const float* __restrict__ W,
                                 unsigned* __restrict__ Wp, int npairs) {
    int i = blockIdx.x * blockDim.x + threadIdx.x;
    if (i >= npairs) return;
    Wp[i] = bf16_bits(W[2 * i]) | (bf16_bits(W[2 * i + 1]) << 16);
}

// ---------------------------------------------------------------------------
// Channel LayerNorm (C == 64), one wave per pixel; optional second LN fused.
// ---------------------------------------------------------------------------
__global__ __launch_bounds__(256)
void ln_kernel(const float* __restrict__ X, float* __restrict__ Y,
               const float* __restrict__ g1, const float* __restrict__ b1,
               const float* __restrict__ g2, const float* __restrict__ b2,
               int dbl) {
    int lane = threadIdx.x & 31;
    int wave = threadIdx.x >> 5;
    int pix  = blockIdx.x * 8 + wave;          // 0..8191
    int bb = pix >> 12, hw = pix & 4095;
    size_t base = ((size_t)bb * 64) * 4096 + hw;
    int c0 = lane, c1 = lane + 32;
    float v0 = X[base + (size_t)c0 * 4096];
    float v1 = X[base + (size_t)c1 * 4096];

    for (int pass = 0; pass < (dbl ? 2 : 1); ++pass) {
        const float* g = pass ? g2 : g1;
        const float* bt = pass ? b2 : b1;
        float s = v0 + v1, ss = v0 * v0 + v1 * v1;
        for (int m = 16; m; m >>= 1) { s += __shfl_xor(s, m); ss += __shfl_xor(ss, m); }
        float mean = s * (1.f / 64.f);
        float var  = ss * (1.f / 64.f) - mean * mean;
        float inv  = rsqrtf(var + 1e-5f);
        v0 = (v0 - mean) * inv * g[c0] + bt[c0];
        v1 = (v1 - mean) * inv * g[c1] + bt[c1];
    }
    Y[base + (size_t)c0 * 4096] = v0;
    Y[base + (size_t)c1 * 4096] = v1;
}

// ---------------------------------------------------------------------------
// Depthwise 3x3 dilated conv (dilation=2, pad=2), groups == C
// ---------------------------------------------------------------------------
__global__ void dwconv_kernel(const float* __restrict__ X, const float* __restrict__ W,
                              const float* __restrict__ Bc, float* __restrict__ Y) {
    int i = blockIdx.x * blockDim.x + threadIdx.x;
    if (i >= 2 * 64 * 4096) return;
    int hw = i & 4095; int t = i >> 12; int ch = t & 63; int b = t >> 6;
    int y = hw >> 6, x = hw & 63;
    float acc = Bc[ch];
    const float* w = W + ch * 9;
    size_t base = ((size_t)(b * 64 + ch)) << 12;
#pragma unroll
    for (int ty = 0; ty < 3; ++ty)
#pragma unroll
        for (int tx = 0; tx < 3; ++tx) {
            int yy = y + (ty - 1) * 2, xx = x + (tx - 1) * 2;
            if (yy >= 0 && yy < 64 && xx >= 0 && xx < 64)
                acc += w[ty * 3 + tx] * X[base + (yy << 6) + xx];
        }
    Y[i] = acc;
}

// ---------------------------------------------------------------------------
// Build 4-direction snake sequences u[b,k,d,l]
// ---------------------------------------------------------------------------
__global__ void build_u_kernel(const float* __restrict__ H, float* __restrict__ U) {
    int i = blockIdx.x * blockDim.x + threadIdx.x;
    if (i >= 2 * 4 * 32 * 4096) return;
    int l = i & 4095; int t = i >> 12;
    int d = t & 31; t >>= 5; int k = t & 3; int b = t >> 2;
    int lsrc = (k & 1) ? 4095 - l : l;
    int f = (k < 2) ? pos_r(lsrc) : idx_c_map(lsrc);
    int ch = (k < 2) ? d : 32 + d;
    U[i] = H[(((size_t)b * 64 + ch) << 12) + f];
}

// x_dbl[b,k,c,l] = sum_d xproj_w[k,c,d] * u[b,k,d,l]    (c in 0..33)
__global__ void xdbl_kernel(const float* __restrict__ U, const float* __restrict__ Wx,
                            float* __restrict__ XD) {
    int i = blockIdx.x * blockDim.x + threadIdx.x;
    if (i >= 2 * 4 * 34 * 4096) return;
    int l = i & 4095; int t = i >> 12;
    int c = t % 34; t /= 34; int k = t & 3; int b = t >> 2;
    const float* up = U + (((size_t)b * 4 + k) * 32) * 4096 + l;
    const float* wp = Wx + (k * 34 + c) * 32;
    float s = 0.f;
#pragma unroll 8
    for (int d = 0; d < 32; ++d) s += wp[d] * up[(size_t)d * 4096];
    XD[i] = s;
}

// delta = softplus(dtproj_w @ dt + dtproj_b), R = 2
__global__ void delta_kernel(const float* __restrict__ XD, const float* __restrict__ Wdt,
                             const float* __restrict__ Bdt, float* __restrict__ DL) {
    int i = blockIdx.x * blockDim.x + threadIdx.x;
    if (i >= 2 * 4 * 32 * 4096) return;
    int l = i & 4095; int t = i >> 12;
    int d = t & 31; t >>= 5; int k = t & 3; int b = t >> 2;
    size_t bx = (((size_t)b * 4 + k) * 34) * 4096;
    float v = Wdt[(k * 32 + d) * 2 + 0] * XD[bx + l] +
              Wdt[(k * 32 + d) * 2 + 1] * XD[bx + 4096 + l] + Bdt[k * 32 + d];
    DL[i] = (v > 20.f) ? v : log1pf(__expf(v));
}

// ---------------------------------------------------------------------------
// Selective scan: one wave handles a (b,k,d-pair); lanes 0-15 -> d0 states,
// lanes 16-31 -> d1 states; cross-lane (16-lane) reduction for y.
// ---------------------------------------------------------------------------
__global__ __launch_bounds__(32)
void scan_kernel(const float* __restrict__ DL, const float* __restrict__ U,
                 const float* __restrict__ XD, const float* __restrict__ Alog,
                 const float* __restrict__ Ds, float* __restrict__ YS) {
    int l = threadIdx.x;
    int blk = blockIdx.x;            // 0..127
    int dp = blk & 15;
    int bk = blk >> 4;
    int b = bk >> 2, k = bk & 3;
    int d = dp * 2 + (l >> 4);
    int n = l & 15;
    float A  = -__expf(Alog[(k * 32 + d) * 16 + n]);
    float Dv = Ds[k * 32 + d];
    size_t baseU = (((size_t)b * 4 + k) * 32 + d) * 4096;
    size_t baseX = (((size_t)b * 4 + k) * 34) * 4096;
    const float* Bp = XD + baseX + (size_t)(2 + n) * 4096;
    const float* Cp = XD + baseX + (size_t)(18 + n) * 4096;
    float h = 0.f;
    for (int t = 0; t < 4096; ++t) {
        if ((t & 63) == 0) {
            __builtin_prefetch(DL + baseU + t + 256, 0, 1);   // global_prefetch_b8
            __builtin_prefetch(U + baseU + t + 256, 0, 1);
        }
        float dl = DL[baseU + t];
        float uv = U[baseU + t];
        float bm = Bp[t], cm = Cp[t];
        h = h * __expf(dl * A) + dl * bm * uv;
        float part = h * cm;
        part += __shfl_xor(part, 1);
        part += __shfl_xor(part, 2);
        part += __shfl_xor(part, 4);
        part += __shfl_xor(part, 8);
        if (n == 0) YS[baseU + t] = part + Dv * uv;
    }
}

// merge 4 directions back to NCHW
__global__ void merge_kernel(const float* __restrict__ YS, float* __restrict__ HM) {
    int i = blockIdx.x * blockDim.x + threadIdx.x;
    if (i >= 2 * 64 * 4096) return;
    int f = i & 4095; int t = i >> 12; int d = t & 63; int b = t >> 6;
    float v;
    if (d < 32) {
        int p = pos_r(f);
        v = YS[(((size_t)b * 4 + 0) * 32 + d) * 4096 + p] +
            YS[(((size_t)b * 4 + 1) * 32 + d) * 4096 + (4095 - p)];
    } else {
        int dd = d - 32, p = pos_c(f);
        v = YS[(((size_t)b * 4 + 2) * 32 + dd) * 4096 + p] +
            YS[(((size_t)b * 4 + 3) * 32 + dd) * 4096 + (4095 - p)];
    }
    HM[i] = v;
}

// ---------------------------------------------------------------------------
// Generic 1x1-conv GEMM on the matrix pipe:
//   Y[co,p] = relu?(W[co,:] . X[:,p] + bias[co]),  P = 8192
// One wave per 16x32 output tile (two 16x16 WMMA tiles sharing the A
// operand), fully unrolled K (bf16 WMMA, f32 accumulate), ping-pong LDS.
// ---------------------------------------------------------------------------
template <int CIN, bool BIAS, bool RELU>
__global__ __launch_bounds__(32)
void gemm_wmma_kernel(const float* __restrict__ X, const unsigned* __restrict__ Wpk,
                      const float* __restrict__ bias, float* __restrict__ Y, int Cout) {
    __shared__ unsigned tile[2][32][16];       // [buf][p][ci-pair] packed bf16
    const int l  = threadIdx.x;
    const int p0 = blockIdx.x * 32;
    const int tc = blockIdx.y;
    const int bidx = p0 >> 12;
    const int hw0  = p0 & 4095;
    const int lh = l >> 4, ll = l & 15;
    constexpr int pitchW = CIN >> 1;
    const int M = tc * 16 + ll;

    v8f acc0 = {}, acc1 = {};
#pragma unroll
    for (int k0 = 0; k0 < CIN; k0 += 32) {
        unsigned (*tl)[16] = tile[(k0 >> 5) & 1];
        // stage B tiles (32 ci x 32 p) into LDS, transposed + bf16-pair packed
        {
            int ci0 = k0 + ll * 2;
            const float* r0 = X + ((size_t)bidx * CIN + ci0) * 4096 + hw0 + lh * 8;
            const float* r1 = r0 + 4096;
#pragma unroll
            for (int j = 0; j < 8; ++j) {
                tl[lh * 8 + j][ll]      = bf16_bits(r0[j])      | (bf16_bits(r1[j])      << 16);
                tl[16 + lh * 8 + j][ll] = bf16_bits(r0[16 + j]) | (bf16_bits(r1[16 + j]) << 16);
            }
        }
        __syncthreads();
        WmmaOp a, b0, b1;
        const unsigned* wrow = Wpk + (size_t)M * pitchW + (k0 >> 1) + lh * 8;
#pragma unroll
        for (int j = 0; j < 8; ++j) { a.u[j] = wrow[j]; }
#pragma unroll
        for (int j = 0; j < 8; ++j) {
            b0.u[j] = tl[ll][lh * 8 + j];
            b1.u[j] = tl[16 + ll][lh * 8 + j];
        }
        acc0 = __builtin_amdgcn_wmma_f32_16x16x32_bf16(
            false, a.v, false, b0.v, (short)0, acc0, false, false);
        acc1 = __builtin_amdgcn_wmma_f32_16x16x32_bf16(
            false, a.v, false, b1.v, (short)0, acc1, false, false);
    }
    const int n = ll, mb = lh * 8;
#pragma unroll
    for (int j = 0; j < 8; ++j) {
        int co = tc * 16 + mb + j;
        float bv = BIAS ? bias[co] : 0.f;
        float v0 = acc0[j] + bv;
        float v1 = acc1[j] + bv;
        if (RELU) { v0 = fmaxf(v0, 0.f); v1 = fmaxf(v1, 0.f); }
        size_t o = ((size_t)bidx * Cout + co) * 4096 + hw0 + n;
        Y[o]      = v0;
        Y[o + 16] = v1;
    }
}

// ---------------------------------------------------------------------------
// BatchNorm statistics: one block per channel; whole channel (2 x 4096 f32)
// pulled into LDS by the Tensor Data Mover, then reduced with ds_add_f32.
// ---------------------------------------------------------------------------
__global__ __launch_bounds__(256)
void bn_stats_kernel(const float* __restrict__ X, int Cn, float* __restrict__ stats) {
    __shared__ float ch[8192];                 // 32 KB tile
    __shared__ float red[2];
    int co = blockIdx.x, tid = threadIdx.x;
    if (tid == 0) { red[0] = 0.f; red[1] = 0.f; }
    if (tid == 0) {
        unsigned long long ga = (unsigned long long)(const void*)(X + (size_t)co * 4096);
        unsigned lds = (unsigned)(size_t)(void*)ch;
        unsigned long long stride0 = (unsigned long long)Cn * 4096ull;  // elements
        uint4v g0;
        g0.x = 1u;                                               // count=1 (valid D#)
        g0.y = lds;                                              // lds_addr
        g0.z = (unsigned)(ga & 0xFFFFFFFFu);                     // global_addr lo
        g0.w = (unsigned)((ga >> 32) & 0x1FFFFFFu) | (2u << 30); // addr hi | type=2
        uint8v g1;
        g1[0] = (2u << 16);                                      // data_size = 4B
        g1[1] = (4096u & 0xFFFFu) << 16;                         // tensor_dim0 lo
        g1[2] = (4096u >> 16) | (2u << 16);                      // dim0 hi | tensor_dim1 lo
        g1[3] = (4096u << 16);                                   // dim1 hi=0 | tile_dim0
        g1[4] = 2u;                                              // tile_dim1=2, tile_dim2=0
        g1[5] = (unsigned)(stride0 & 0xFFFFFFFFu);               // dim0_stride lo
        g1[6] = (unsigned)((stride0 >> 32) & 0xFFFFu);           // dim0_stride hi
        g1[7] = 0u;
        asm volatile("tensor_load_to_lds %0, %1" :: "s"(g0), "s"(g1) : "memory");
        __builtin_amdgcn_s_wait_tensorcnt(0);
    }
    __syncthreads();
    float s = 0.f, ss = 0.f;
    for (int j = tid; j < 8192; j += 256) { float v = ch[j]; s += v; ss += v * v; }
    for (int m = 16; m; m >>= 1) { s += __shfl_xor(s, m); ss += __shfl_xor(ss, m); }
    if ((tid & 31) == 0) { atomicAdd(&red[0], s); atomicAdd(&red[1], ss); }
    __syncthreads();
    if (tid == 0) {
        float mean = red[0] * (1.f / 8192.f);
        float var  = red[1] * (1.f / 8192.f) - mean * mean;
        stats[co * 2 + 0] = mean;
        stats[co * 2 + 1] = rsqrtf(var + 1e-5f);
    }
}

__global__ void bn_apply_kernel(const float* __restrict__ X, const float* __restrict__ stats,
                                const float* __restrict__ g, const float* __restrict__ b,
                                const float* __restrict__ resid, float* __restrict__ Y,
                                int Cn, int total) {
    int i = blockIdx.x * blockDim.x + threadIdx.x;
    if (i >= total) return;
    int ch = (i >> 12) % Cn;
    float v = (X[i] - stats[ch * 2]) * stats[ch * 2 + 1] * g[ch] + b[ch];
    if (resid) v += resid[i];
    Y[i] = v;
}

// per-(b,c) spatial mean for ECA; row staged through LDS with async copies
// (GLOBAL_LOAD_ASYNC_TO_LDS_B128, tracked by ASYNCcnt)
__global__ __launch_bounds__(256)
void chmean_kernel(const float* __restrict__ X, float* __restrict__ out) {
    __shared__ float buf[4096];
    __shared__ float red;
    int tid = threadIdx.x;
    if (tid == 0) red = 0.f;
    size_t base = (size_t)blockIdx.x * 4096;
#pragma unroll
    for (int c = 0; c < 4; ++c) {
        int e = (c * 256 + tid) * 4;
        unsigned ldsoff = (unsigned)(size_t)(const void*)(buf + e);
        unsigned long long ga = (unsigned long long)(const void*)(X + base + e);
        asm volatile("global_load_async_to_lds_b128 %0, %1, off"
                     :: "v"(ldsoff), "v"(ga) : "memory");
    }
    asm volatile("s_wait_asynccnt 0" ::: "memory");
    __syncthreads();
    float s = 0.f;
    for (int j = tid; j < 4096; j += 256) s += buf[j];
    for (int m = 16; m; m >>= 1) s += __shfl_xor(s, m);
    if ((tid & 31) == 0) atomicAdd(&red, s);
    __syncthreads();
    if (tid == 0) out[blockIdx.x] = red * (1.f / 4096.f);
}

// ECA gate + residual:  x1 = p * sigmoid(conv1d_k3(mean)) + shortcut
__global__ void eca_apply_kernel(const float* __restrict__ P, const float* __restrict__ mbc,
                                 const float* __restrict__ w, const float* __restrict__ xin,
                                 float* __restrict__ X1) {
    int i = blockIdx.x * blockDim.x + threadIdx.x;
    if (i >= 2 * 64 * 4096) return;
    int t = i >> 12; int ch = t & 63;
    float m0 = mbc[t];
    float mm = (ch > 0)  ? mbc[t - 1] : 0.f;
    float mp = (ch < 63) ? mbc[t + 1] : 0.f;
    float y  = w[0] * mm + w[1] * m0 + w[2] * mp;
    float sg = 1.f / (1.f + __expf(-y));
    X1[i] = P[i] * sg + xin[i];
}

// mixed-kernel depthwise convs (1/3/5/7) + residual + relu on HID=256 channels
__global__ void dwmix_kernel(const float* __restrict__ H,
                             const float* __restrict__ w1, const float* __restrict__ b1,
                             const float* __restrict__ w3, const float* __restrict__ b3,
                             const float* __restrict__ w5, const float* __restrict__ b5,
                             const float* __restrict__ w7, const float* __restrict__ b7,
                             float* __restrict__ Y) {
    int i = blockIdx.x * blockDim.x + threadIdx.x;
    if (i >= 2 * 256 * 4096) return;
    int hw = i & 4095; int t = i >> 12; int co = t & 255; int b = t >> 8;
    int gi = co >> 6, cc = co & 63;
    int kk = 1 + 2 * gi, pad = gi;
    const float* w; const float* bs;
    if (gi == 0) { w = w1; bs = b1; }
    else if (gi == 1) { w = w3; bs = b3; }
    else if (gi == 2) { w = w5; bs = b5; }
    else { w = w7; bs = b7; }
    int y = hw >> 6, x = hw & 63;
    size_t base = ((size_t)(b * 256 + co)) << 12;
    float acc = bs[cc];
    for (int ky = 0; ky < kk; ++ky)
        for (int kx = 0; kx < kk; ++kx) {
            int yy = y + ky - pad, xx = x + kx - pad;
            if (yy >= 0 && yy < 64 && xx >= 0 && xx < 64)
                acc += w[cc * kk * kk + ky * kk + kx] * H[base + (yy << 6) + xx];
        }
    acc += H[i];                 // residual (+ hm)
    Y[i] = fmaxf(acc, 0.f);      // relu before BN2
}

// ---------------------------------------------------------------------------
extern "C" void kernel_launch(void* const* d_in, const int* in_sizes, int n_in,
                              void* d_out, int out_size, void* d_ws, size_t ws_size,
                              hipStream_t stream) {
    (void)in_sizes; (void)n_in; (void)out_size; (void)ws_size;
    const float* x       = (const float*)d_in[0];
    const float* ln1_g   = (const float*)d_in[1];
    const float* ln1_b   = (const float*)d_in[2];
    const float* ln2_g   = (const float*)d_in[3];
    const float* ln2_b   = (const float*)d_in[4];
    const float* con1_w  = (const float*)d_in[5];
    const float* con1_b  = (const float*)d_in[6];
    const float* xproj_w = (const float*)d_in[7];
    const float* dtproj_w= (const float*)d_in[8];
    const float* dtproj_b= (const float*)d_in[9];
    const float* A_log   = (const float*)d_in[10];
    const float* Ds      = (const float*)d_in[11];
    const float* ssm_g   = (const float*)d_in[12];
    const float* ssm_b   = (const float*)d_in[13];
    const float* proj_w  = (const float*)d_in[14];
    const float* proj_b  = (const float*)d_in[15];
    const float* pbn_g   = (const float*)d_in[16];
    const float* pbn_b   = (const float*)d_in[17];
    const float* eca_w   = (const float*)d_in[18];
    const float* fc1_w   = (const float*)d_in[19];
    const float* bn1_g   = (const float*)d_in[20];
    const float* bn1_b   = (const float*)d_in[21];
    const float* dw_w1   = (const float*)d_in[22];
    const float* dw_b1   = (const float*)d_in[23];
    const float* dw_w3   = (const float*)d_in[24];
    const float* dw_b3   = (const float*)d_in[25];
    const float* dw_w5   = (const float*)d_in[26];
    const float* dw_b5   = (const float*)d_in[27];
    const float* dw_w7   = (const float*)d_in[28];
    const float* dw_b7   = (const float*)d_in[29];
    const float* bn2_g   = (const float*)d_in[30];
    const float* bn2_b   = (const float*)d_in[31];
    const float* fc2_w   = (const float*)d_in[32];
    const float* bn3_g   = (const float*)d_in[33];
    const float* bn3_b   = (const float*)d_in[34];

    char* ws = (char*)d_ws;
    size_t off = 0;
    auto alloc = [&](size_t bytes) { size_t r = off; off += (bytes + 255) & ~(size_t)255; return r; };
    const size_t F = sizeof(float);
    unsigned* wpk_proj = (unsigned*)(ws + alloc(64 * 32 * 4));
    unsigned* wpk_fc1  = (unsigned*)(ws + alloc(256 * 32 * 4));
    unsigned* wpk_fc2  = (unsigned*)(ws + alloc(64 * 128 * 4));
    float* hln   = (float*)(ws + alloc(2 * 64 * 4096 * F));
    float* hcv   = (float*)(ws + alloc(2 * 64 * 4096 * F));
    float* ubuf  = (float*)(ws + alloc(2 * 4 * 32 * 4096 * F));
    float* xdbl  = (float*)(ws + alloc(2 * 4 * 34 * 4096 * F));
    float* dlt   = (float*)(ws + alloc(2 * 4 * 32 * 4096 * F));
    float* yscan = (float*)(ws + alloc(2 * 4 * 32 * 4096 * F));
    float* hmerge= (float*)(ws + alloc(2 * 64 * 4096 * F));
    float* hpost = (float*)(ws + alloc(2 * 64 * 4096 * F));
    float* pbuf  = (float*)(ws + alloc(2 * 64 * 4096 * F));
    float* pnorm = (float*)(ws + alloc(2 * 64 * 4096 * F));
    float* x1    = (float*)(ws + alloc(2 * 64 * 4096 * F));
    float* mbuf  = (float*)(ws + alloc(2 * 64 * 4096 * F));
    float* g1    = (float*)(ws + alloc(2 * 256 * 4096 * F));
    float* hmbuf = (float*)(ws + alloc(2 * 256 * 4096 * F));
    float* g2    = (float*)(ws + alloc(2 * 256 * 4096 * F));
    float* hm2   = (float*)(ws + alloc(2 * 256 * 4096 * F));
    float* fbuf  = (float*)(ws + alloc(2 * 64 * 4096 * F));
    float* stats = (float*)(ws + alloc(256 * 2 * F));
    float* mbc   = (float*)(ws + alloc(128 * F));
    float* outp  = (float*)d_out;

    const int T = 256;
    // weight packing
    pack_bf16_kernel<<<(2048 + T - 1) / T, T, 0, stream>>>(proj_w, wpk_proj, 2048);
    pack_bf16_kernel<<<(8192 + T - 1) / T, T, 0, stream>>>(fc1_w, wpk_fc1, 8192);
    pack_bf16_kernel<<<(8192 + T - 1) / T, T, 0, stream>>>(fc2_w, wpk_fc2, 8192);

    // ln1 -> depthwise dilated conv
    ln_kernel<<<1024, 256, 0, stream>>>(x, hln, ln1_g, ln1_b, nullptr, nullptr, 0);
    dwconv_kernel<<<(524288 + T - 1) / T, T, 0, stream>>>(hln, con1_w, con1_b, hcv);

    // snake SS2D
    build_u_kernel<<<(1048576 + T - 1) / T, T, 0, stream>>>(hcv, ubuf);
    xdbl_kernel<<<(1114112 + T - 1) / T, T, 0, stream>>>(ubuf, xproj_w, xdbl);
    delta_kernel<<<(1048576 + T - 1) / T, T, 0, stream>>>(xdbl, dtproj_w, dtproj_b, dlt);
    scan_kernel<<<128, 32, 0, stream>>>(dlt, ubuf, xdbl, A_log, Ds, yscan);
    merge_kernel<<<(524288 + T - 1) / T, T, 0, stream>>>(yscan, hmerge);
    ln_kernel<<<1024, 256, 0, stream>>>(hmerge, hpost, ssm_g, ssm_b, ln2_g, ln2_b, 1);

    // proj 1x1 conv (WMMA) + relu -> BN -> ECA + shortcut
    gemm_wmma_kernel<64, true, true><<<dim3(256, 4), 32, 0, stream>>>(
        hpost, wpk_proj, proj_b, pbuf, 64);
    bn_stats_kernel<<<64, 256, 0, stream>>>(pbuf, 64, stats);
    bn_apply_kernel<<<(524288 + T - 1) / T, T, 0, stream>>>(pbuf, stats, pbn_g, pbn_b,
                                                            nullptr, pnorm, 64, 524288);
    chmean_kernel<<<128, 256, 0, stream>>>(pnorm, mbc);
    eca_apply_kernel<<<(524288 + T - 1) / T, T, 0, stream>>>(pnorm, mbc, eca_w, x, x1);

    // MLP branch: LN -> fc1 (WMMA, relu) -> BN1 -> dwmix -> BN2 -> fc2 (WMMA) -> BN3 + x1
    ln_kernel<<<1024, 256, 0, stream>>>(x1, mbuf, ln2_g, ln2_b, nullptr, nullptr, 0);
    gemm_wmma_kernel<64, false, true><<<dim3(256, 16), 32, 0, stream>>>(
        mbuf, wpk_fc1, nullptr, g1, 256);
    bn_stats_kernel<<<256, 256, 0, stream>>>(g1, 256, stats);
    bn_apply_kernel<<<(2097152 + T - 1) / T, T, 0, stream>>>(g1, stats, bn1_g, bn1_b,
                                                             nullptr, hmbuf, 256, 2097152);
    dwmix_kernel<<<(2097152 + T - 1) / T, T, 0, stream>>>(hmbuf, dw_w1, dw_b1, dw_w3, dw_b3,
                                                          dw_w5, dw_b5, dw_w7, dw_b7, g2);
    bn_stats_kernel<<<256, 256, 0, stream>>>(g2, 256, stats);
    bn_apply_kernel<<<(2097152 + T - 1) / T, T, 0, stream>>>(g2, stats, bn2_g, bn2_b,
                                                             nullptr, hm2, 256, 2097152);
    gemm_wmma_kernel<256, false, false><<<dim3(256, 4), 32, 0, stream>>>(
        hm2, wpk_fc2, nullptr, fbuf, 64);
    bn_stats_kernel<<<64, 256, 0, stream>>>(fbuf, 64, stats);
    bn_apply_kernel<<<(524288 + T - 1) / T, T, 0, stream>>>(fbuf, stats, bn3_g, bn3_b,
                                                            x1, outp, 64, 524288);
}
